// SelfAttention_34754875359745
// MI455X (gfx1250) — matrix operand order repaired and checked by reading, not compiled, not run on hardware
//
#include <hip/hip_runtime.h>
#include <cstdint>

typedef __attribute__((ext_vector_type(16))) _Float16 v16h;
typedef __attribute__((ext_vector_type(8)))  _Float16 v8h;
typedef __attribute__((ext_vector_type(8)))  float    v8f;
typedef __attribute__((ext_vector_type(4)))  int      v4i;

#define DIM 768
#define NHEAD 12
#define HDIM 64
#define SEQ 1024
#define BATCH 8
#define ROWS (BATCH * SEQ)   // 8192

// ---------------------------------------------------------------------------
// gfx1250 async global->LDS copy (ASYNCcnt-tracked). The builtin takes
// (v4i AS1* gaddr, v4i AS3* ldsaddr, imm offset, imm cpol) — confirmed by the
// round-2 diagnostic (AS1 prints as `__device__` in HIP mode). Generic LDS
// addresses carry the byte offset in bits [31:0], so the integer-truncation
// cast to AS(3) is address-correct.
// ---------------------------------------------------------------------------
#if __has_builtin(__builtin_amdgcn_global_load_async_to_lds_b128)
#define HAVE_ASYNC 1
typedef __attribute__((address_space(1))) v4i g_v4i;
typedef __attribute__((address_space(3))) v4i l_v4i;
__device__ __forceinline__ void async_copy16(const _Float16* g, _Float16* l) {
  __builtin_amdgcn_global_load_async_to_lds_b128(
      (g_v4i*)(uintptr_t)g, (l_v4i*)(unsigned)(uintptr_t)l, 0, 0);
}
#else
#define HAVE_ASYNC 0
__device__ __forceinline__ void async_copy16(const _Float16* g, _Float16* l) {
  *(v8h*)l = *(const v8h*)g;
}
#endif

#if __has_builtin(__builtin_amdgcn_s_wait_asynccnt)
#define WAIT_ASYNC() __builtin_amdgcn_s_wait_asynccnt(0)
#else
#define WAIT_ASYNC() asm volatile("s_wait_asynccnt 0x0" ::: "memory")
#endif

// ---------------------------------------------------------------------------
// WMMA helpers (CDNA5 wave32, V_WMMA_F32_16X16X32_F16)
// ---------------------------------------------------------------------------
__device__ __forceinline__ v8f wmma_f16(v16h a, v16h b, v8f c) {
  return __builtin_amdgcn_wmma_f32_16x16x32_f16(
      /*neg_a=*/false, a, /*neg_b=*/false, b,
      /*c_mod=*/(short)0, c, /*reuse_a=*/false, /*reuse_b=*/false);
}

// A operand: 16(M) x 32(K) f16, row-major, leading dim ldm.
// lanes 0-15 -> M=lane,   halves [0..7]=K0..7,  [8..15]=K16..23
// lanes 16-31-> M=lane-16, halves [0..7]=K8..15, [8..15]=K24..31
__device__ __forceinline__ v16h loadA16x32(const _Float16* p, int ldm, int lane) {
  const int m  = lane & 15;
  const int kb = (lane < 16) ? 0 : 8;
  const _Float16* r = p + m * ldm;
  v16h a;
#pragma unroll
  for (int i = 0; i < 8; ++i) a[i] = r[kb + i];
#pragma unroll
  for (int i = 0; i < 8; ++i) a[8 + i] = r[kb + 16 + i];
  return a;
}

// B operand: 32(K) x 16(N) f16, stored K-contiguous per column: (k,n) at p[n*ldn+k].
// lanes 0-15 -> N=lane, K 0..7,16..23; lanes 16-31 -> N=lane-16, K 8..15,24..31
__device__ __forceinline__ v16h loadB32x16(const _Float16* p, int ldn, int lane) {
  const int n  = lane & 15;
  const int kb = (lane < 16) ? 0 : 8;
  const _Float16* r = p + n * ldn;
  v16h b;
#pragma unroll
  for (int i = 0; i < 8; ++i) { b[i] = r[kb + i]; b[8 + i] = r[kb + 16 + i]; }
  return b;
}

// ---------------------------------------------------------------------------
// K0: f32 -> f16 conversion (weights)
// ---------------------------------------------------------------------------
__global__ void cvt_f16_kernel(const float* __restrict__ src,
                               _Float16* __restrict__ dst, int n) {
  int i = blockIdx.x * blockDim.x + threadIdx.x;
  if (i < n) dst[i] = (_Float16)src[i];
}

// ---------------------------------------------------------------------------
// K1: LayerNorm, one wave per row of 768, f16 output
// ---------------------------------------------------------------------------
__global__ __launch_bounds__(256) void ln_kernel(
    const float* __restrict__ x, const float* __restrict__ gamma,
    const float* __restrict__ beta, _Float16* __restrict__ xn) {
  const int lane = threadIdx.x & 31;
  const int wv   = threadIdx.x >> 5;
  const int row  = blockIdx.x * 8 + wv;
  const float* xr = x + (size_t)row * DIM;
  float v[24];
  float s = 0.f, ss = 0.f;
#pragma unroll
  for (int i = 0; i < 24; ++i) {
    v[i] = xr[lane + i * 32];
    s  += v[i];
    ss += v[i] * v[i];
  }
#pragma unroll
  for (int m = 16; m >= 1; m >>= 1) {
    s  += __shfl_xor(s,  m, 32);
    ss += __shfl_xor(ss, m, 32);
  }
  const float mu   = s * (1.0f / DIM);
  const float var  = ss * (1.0f / DIM) - mu * mu;
  const float rstd = rsqrtf(var + 1e-5f);
  _Float16* xo = xn + (size_t)row * DIM;
#pragma unroll
  for (int i = 0; i < 24; ++i) {
    const int c = lane + i * 32;
    xo[c] = (_Float16)((v[i] - mu) * rstd * gamma[c] + beta[c]);
  }
}

// ---------------------------------------------------------------------------
// Shared GEMM mainloop: block 256 thr = 8 waves, tile 256(M) x 64(N), K-step 32,
// wave tile 32x64 (8 WMMA / K-step), double-buffered LDS with async A prefetch.
// ---------------------------------------------------------------------------
#define LDA 40   // 32 + 8 halves
#define LDB 72   // 64 + 8 halves
#define KSTEPS (DIM / 32)

__device__ __forceinline__ void gemm_mainloop(
    const _Float16* __restrict__ Amat, const _Float16* __restrict__ Wmat,
    int row0, int n0, _Float16 (*Ah)[256 * LDA], _Float16 (*Bt)[64 * LDA],
    v8f acc[2][4]) {
  const int tid  = threadIdx.x;
  const int lane = tid & 31;
  const int wv   = tid >> 5;
  const int br   = tid >> 3;        // B stage: row k
  const int bc   = (tid & 7) * 8;   // B stage: col group

  auto stageA = [&](int buf, int k0) {
#pragma unroll
    for (int i = 0; i < 4; ++i) {
      const int u = tid + i * 256;           // 1024 16-byte units
      const int r = u >> 2;
      const int c = (u & 3) * 8;
      async_copy16(&Amat[(size_t)(row0 + r) * DIM + k0 + c], &Ah[buf][r * LDA + c]);
    }
  };
  auto loadBreg = [&](int k0) -> v8h {
    return *(const v8h*)&Wmat[(size_t)(k0 + br) * DIM + n0 + bc];
  };
  auto scatterB = [&](int buf, v8h w) {
#pragma unroll
    for (int j = 0; j < 8; ++j) Bt[buf][(bc + j) * LDA + br] = w[j];
  };
  auto compute = [&](int buf) {
    const v16h a0 = loadA16x32(&Ah[buf][(wv * 32) * LDA], LDA, lane);
    const v16h a1 = loadA16x32(&Ah[buf][(wv * 32 + 16) * LDA], LDA, lane);
    v16h b[4];
#pragma unroll
    for (int nt = 0; nt < 4; ++nt)
      b[nt] = loadB32x16(&Bt[buf][(nt * 16) * LDA], LDA, lane);
#pragma unroll
    for (int nt = 0; nt < 4; ++nt) acc[0][nt] = wmma_f16(a0, b[nt], acc[0][nt]);
#pragma unroll
    for (int nt = 0; nt < 4; ++nt) acc[1][nt] = wmma_f16(a1, b[nt], acc[1][nt]);
  };

  stageA(0, 0);
  scatterB(0, loadBreg(0));
  WAIT_ASYNC();
  __syncthreads();
  for (int kt = 0; kt < KSTEPS; ++kt) {
    const int  cur  = kt & 1;
    const bool more = (kt + 1) < KSTEPS;
    v8h wreg;
    if (more) {
      stageA(cur ^ 1, (kt + 1) * 32);       // async prefetch overlaps compute
      wreg = loadBreg((kt + 1) * 32);
    }
    compute(cur);
    if (more) scatterB(cur ^ 1, wreg);
    WAIT_ASYNC();
    __syncthreads();
  }
}

// ---------------------------------------------------------------------------
// K2: projection GEMM, scatter to (B,H,N,64) f16, +bias (+pos for Q)
// ---------------------------------------------------------------------------
__global__ __launch_bounds__(256) void qkv_gemm_kernel(
    const _Float16* __restrict__ xn, const _Float16* __restrict__ Wh,
    const float* __restrict__ bias, const float* __restrict__ pos, int addPos,
    _Float16* __restrict__ out) {
  __shared__ _Float16 Ah[2][256 * LDA];
  __shared__ _Float16 Bt[2][64 * LDA];
  const int lane = threadIdx.x & 31;
  const int wv   = threadIdx.x >> 5;
  const int n0   = blockIdx.x * 64;
  const int row0 = blockIdx.y * 256;

  v8f acc[2][4] = {};
  gemm_mainloop(xn, Wh, row0, n0, Ah, Bt, acc);

  const int nl   = lane & 15;
  const int mofs = (lane < 16) ? 0 : 8;
  const int h    = blockIdx.x;   // N-tile == one head
#pragma unroll
  for (int hf = 0; hf < 2; ++hf) {
#pragma unroll
    for (int nt = 0; nt < 4; ++nt) {
#pragma unroll
      for (int r = 0; r < 8; ++r) {
        const int gm = row0 + wv * 32 + hf * 16 + mofs + r;
        const int gc = n0 + nt * 16 + nl;
        float vv = acc[hf][nt][r] + bias[gc];
        if (addPos) vv += pos[(size_t)gm * DIM + gc];
        const int b = gm >> 10, nseq = gm & 1023;
        const int d = gc & 63;
        out[(((size_t)b * NHEAD + h) * SEQ + nseq) * HDIM + d] = (_Float16)vv;
      }
    }
  }
}

// ---------------------------------------------------------------------------
// K3: flash attention. grid (8 qblocks, 96 bh). block 256 = 8 waves,
// wave owns 16 query rows; double-buffered 32-key tiles; online softmax.
// ---------------------------------------------------------------------------
#define LDV 40
#define LDP 40
__global__ __launch_bounds__(256) void attn_kernel(
    const _Float16* __restrict__ Qh, const _Float16* __restrict__ Kh,
    const _Float16* __restrict__ Vh, _Float16* __restrict__ Oh) {
  __shared__ _Float16 Ks[2][32 * LDB];     // K tile natural [key][d]
  __shared__ _Float16 Vt[2][64 * LDV];     // V tile transposed [d][key]
  __shared__ _Float16 Ps[8 * 16 * LDP];    // per-wave 16x32 probabilities
  const int tid  = threadIdx.x;
  const int lane = tid & 31;
  const int wv   = tid >> 5;
  const int bh   = blockIdx.y;
  const int q0   = blockIdx.x * 128 + wv * 16;
  const _Float16* Qb = Qh + (size_t)bh * SEQ * HDIM;
  const _Float16* Kb = Kh + (size_t)bh * SEQ * HDIM;
  const _Float16* Vb = Vh + (size_t)bh * SEQ * HDIM;

  // Q strip -> two A operands (d 0..31 and 32..63), loaded once
  v16h aq0, aq1;
  {
    const int m  = lane & 15;
    const int kb = (lane < 16) ? 0 : 8;
    const _Float16* qr = Qb + (size_t)(q0 + m) * HDIM;
#pragma unroll
    for (int i = 0; i < 8; ++i) { aq0[i] = qr[kb + i];      aq0[8 + i] = qr[kb + 16 + i]; }
#pragma unroll
    for (int i = 0; i < 8; ++i) { aq1[i] = qr[32 + kb + i]; aq1[8 + i] = qr[32 + kb + 16 + i]; }
  }

  const int vr = tid >> 3;        // staging row (key)
  const int vc = (tid & 7) * 8;   // staging col group (d)
  auto stageK = [&](int buf, int j0) {
    async_copy16(&Kb[(size_t)(j0 + vr) * HDIM + vc], &Ks[buf][vr * LDB + vc]);
  };
  auto loadVreg = [&](int j0) -> v8h {
    return *(const v8h*)&Vb[(size_t)(j0 + vr) * HDIM + vc];
  };
  auto scatterV = [&](int buf, v8h vv) {
#pragma unroll
    for (int t = 0; t < 8; ++t) Vt[buf][(vc + t) * LDV + vr] = vv[t];
  };

  float mrun[8], lrun[8];
#pragma unroll
  for (int r = 0; r < 8; ++r) { mrun[r] = -1e30f; lrun[r] = 0.f; }
  v8f o[4] = {};
  const float scale = 0.125f;   // 1/sqrt(64)
  const int mrow = (lane < 16) ? 0 : 8;
  _Float16* Pw = &Ps[wv * 16 * LDP];

  stageK(0, 0);
  scatterV(0, loadVreg(0));
  WAIT_ASYNC();
  __syncthreads();

  for (int jt = 0; jt < SEQ / 32; ++jt) {
    const int  cur  = jt & 1;
    const bool more = (jt + 1) < SEQ / 32;
    v8h vreg;
    if (more) {
      stageK(cur ^ 1, (jt + 1) * 32);
      vreg = loadVreg((jt + 1) * 32);
    }

    // S = Q K^T : all operand loads first, then 4 back-to-back WMMAs
    v16h bk[4];
    bk[0] = loadB32x16(&Ks[cur][0],            LDB, lane);  // keys 0..15, d 0..31
    bk[1] = loadB32x16(&Ks[cur][32],           LDB, lane);  // keys 0..15, d 32..63
    bk[2] = loadB32x16(&Ks[cur][16 * LDB],      LDB, lane); // keys 16..31
    bk[3] = loadB32x16(&Ks[cur][16 * LDB + 32], LDB, lane);
    v8f s0 = {}, s1 = {};
    s0 = wmma_f16(aq0, bk[0], s0);
    s0 = wmma_f16(aq1, bk[1], s0);
    s1 = wmma_f16(aq0, bk[2], s1);
    s1 = wmma_f16(aq1, bk[3], s1);

    // online softmax: each C-tile row lives in one VGPR across a 16-lane
    // half-group, so xor masks 1..8 stay inside the group.
    float al[8];
#pragma unroll
    for (int r = 0; r < 8; ++r) {
      const float x0 = s0[r] * scale;
      const float x1 = s1[r] * scale;
      float mx = fmaxf(x0, x1);
#pragma unroll
      for (int msk = 8; msk >= 1; msk >>= 1) mx = fmaxf(mx, __shfl_xor(mx, msk, 32));
      const float mnew = fmaxf(mrun[r], mx);
      const float p0 = __expf(x0 - mnew);
      const float p1 = __expf(x1 - mnew);
      al[r] = __expf(mrun[r] - mnew);
      float rs = p0 + p1;
#pragma unroll
      for (int msk = 8; msk >= 1; msk >>= 1) rs += __shfl_xor(rs, msk, 32);
      lrun[r] = lrun[r] * al[r] + rs;
      mrun[r] = mnew;
      _Float16* pr = &Pw[(r + mrow) * LDP];     // C-layout -> A-layout via LDS
      pr[lane & 15]        = (_Float16)p0;
      pr[16 + (lane & 15)] = (_Float16)p1;
    }
#pragma unroll
    for (int nt = 0; nt < 4; ++nt)
#pragma unroll
      for (int r = 0; r < 8; ++r) o[nt][r] *= al[r];

    // O += P @ V : operand loads first, then 4 WMMAs
    const v16h ap = loadA16x32(Pw, LDP, lane);
    v16h bv[4];
#pragma unroll
    for (int nt = 0; nt < 4; ++nt)
      bv[nt] = loadB32x16(&Vt[cur][(nt * 16) * LDV], LDV, lane);
#pragma unroll
    for (int nt = 0; nt < 4; ++nt) o[nt] = wmma_f16(ap, bv[nt], o[nt]);

    if (more) scatterV(cur ^ 1, vreg);
    WAIT_ASYNC();
    __syncthreads();
  }

  // normalize and scatter to (B, N, 768) f16 for the output GEMM
  const int nl = lane & 15;
  const int b  = bh / NHEAD, h = bh % NHEAD;
#pragma unroll
  for (int nt = 0; nt < 4; ++nt) {
#pragma unroll
    for (int r = 0; r < 8; ++r) {
      const float vv = o[nt][r] / lrun[r];
      const int q = q0 + mrow + r;
      const int d = nt * 16 + nl;
      Oh[((size_t)b * SEQ + q) * DIM + h * HDIM + d] = (_Float16)vv;
    }
  }
}

// ---------------------------------------------------------------------------
// K4: output projection + bias + residual, fp32 out
// ---------------------------------------------------------------------------
__global__ __launch_bounds__(256) void out_gemm_kernel(
    const _Float16* __restrict__ Ain, const _Float16* __restrict__ Wh,
    const float* __restrict__ bias, const float* __restrict__ resid,
    float* __restrict__ out) {
  __shared__ _Float16 Ah[2][256 * LDA];
  __shared__ _Float16 Bt[2][64 * LDA];
  const int lane = threadIdx.x & 31;
  const int wv   = threadIdx.x >> 5;
  const int n0   = blockIdx.x * 64;
  const int row0 = blockIdx.y * 256;

  v8f acc[2][4] = {};
  gemm_mainloop(Ain, Wh, row0, n0, Ah, Bt, acc);

  const int nl   = lane & 15;
  const int mofs = (lane < 16) ? 0 : 8;
#pragma unroll
  for (int hf = 0; hf < 2; ++hf) {
#pragma unroll
    for (int nt = 0; nt < 4; ++nt) {
#pragma unroll
      for (int r = 0; r < 8; ++r) {
        const int gm = row0 + wv * 32 + hf * 16 + mofs + r;
        const int gc = n0 + nt * 16 + nl;
        out[(size_t)gm * DIM + gc] =
            acc[hf][nt][r] + bias[gc] + resid[(size_t)gm * DIM + gc];
      }
    }
  }
}

// ---------------------------------------------------------------------------
// launch
// ---------------------------------------------------------------------------
extern "C" void kernel_launch(void* const* d_in, const int* in_sizes, int n_in,
                              void* d_out, int out_size, void* d_ws, size_t ws_size,
                              hipStream_t stream) {
  const float* x     = (const float*)d_in[0];
  const float* pos   = (const float*)d_in[1];
  const float* gamma = (const float*)d_in[2];
  const float* beta  = (const float*)d_in[3];
  const float* Wq = (const float*)d_in[4];
  const float* bq = (const float*)d_in[5];
  const float* Wk = (const float*)d_in[6];
  const float* bk = (const float*)d_in[7];
  const float* Wv = (const float*)d_in[8];
  const float* bv = (const float*)d_in[9];
  const float* Wo = (const float*)d_in[10];
  const float* bo = (const float*)d_in[11];
  float* out = (float*)d_out;

  const size_t WN = (size_t)DIM * DIM;
  const size_t XN = (size_t)ROWS * DIM;
  char* ws = (char*)d_ws;
  _Float16* Wq_h = (_Float16*)ws;  ws += WN * 2;
  _Float16* Wk_h = (_Float16*)ws;  ws += WN * 2;
  _Float16* Wv_h = (_Float16*)ws;  ws += WN * 2;
  _Float16* Wo_h = (_Float16*)ws;  ws += WN * 2;
  _Float16* xn_h = (_Float16*)ws;  ws += XN * 2;
  _Float16* Q_h  = (_Float16*)ws;  ws += XN * 2;
  _Float16* K_h  = (_Float16*)ws;  ws += XN * 2;
  _Float16* V_h  = (_Float16*)ws;  ws += XN * 2;
  _Float16* A_h  = (_Float16*)ws;  ws += XN * 2;

  {
    const int n = (int)WN, blk = 256, grd = (n + blk - 1) / blk;
    cvt_f16_kernel<<<grd, blk, 0, stream>>>(Wq, Wq_h, n);
    cvt_f16_kernel<<<grd, blk, 0, stream>>>(Wk, Wk_h, n);
    cvt_f16_kernel<<<grd, blk, 0, stream>>>(Wv, Wv_h, n);
    cvt_f16_kernel<<<grd, blk, 0, stream>>>(Wo, Wo_h, n);
  }
  ln_kernel<<<ROWS / 8, 256, 0, stream>>>(x, gamma, beta, xn_h);
  {
    dim3 grd(DIM / 64, ROWS / 256);
    qkv_gemm_kernel<<<grd, 256, 0, stream>>>(xn_h, Wq_h, bq, pos, 1, Q_h);
    qkv_gemm_kernel<<<grd, 256, 0, stream>>>(xn_h, Wk_h, bk, pos, 0, K_h);
    qkv_gemm_kernel<<<grd, 256, 0, stream>>>(xn_h, Wv_h, bv, pos, 0, V_h);
  }
  attn_kernel<<<dim3(SEQ / 128, BATCH * NHEAD), 256, 0, stream>>>(Q_h, K_h, V_h, A_h);
  out_gemm_kernel<<<dim3(DIM / 64, ROWS / 256), 256, 0, stream>>>(A_h, Wo_h, bo, x, out);
}
